// FitTorch_87239375716512
// MI455X (gfx1250) — matrix operand order repaired and checked by reading, not compiled, not run on hardware
//
#include <hip/hip_runtime.h>
#include <math.h>
#include <stdint.h>

typedef __attribute__((ext_vector_type(16))) _Float16 v16h;
typedef __attribute__((ext_vector_type(8)))  _Float16 v8h;
typedef __attribute__((ext_vector_type(8)))  float    v8f;
typedef unsigned int u32;
typedef __attribute__((ext_vector_type(4))) u32 u32x4;
typedef __attribute__((ext_vector_type(8))) int i32x8;
typedef __attribute__((ext_vector_type(4))) int i32x4;

#define HIDDEN 64
#define DEG 32

// Per-wave LDS fence: all 32 lanes run in lockstep; waiting DScnt==0 makes
// prior same-wave LDS writes visible to subsequent cross-lane LDS reads.
#define LDS_FENCE() asm volatile("s_wait_dscnt 0" ::: "memory")

#if __has_builtin(__builtin_amdgcn_rcpf)
#define FRCP(x) __builtin_amdgcn_rcpf(x)
#else
#define FRCP(x) (1.0f / (x))
#endif

#if __has_builtin(__builtin_amdgcn_tensor_load_to_lds) && \
    __has_builtin(__builtin_amdgcn_s_wait_tensorcnt)
#define HAVE_TDM 1
#else
#define HAVE_TDM 0
#endif

__device__ __forceinline__ float sigf(float z) {
  return FRCP(1.0f + __expf(-z));  // v_exp_f32 + v_rcp_f32, no IEEE divide
}
__device__ __forceinline__ float dsilu(float z, float s) {
  return s * (1.0f + z * (1.0f - s));
}

__device__ __forceinline__ v8f vzero8() {
  v8f z;
#pragma unroll
  for (int i = 0; i < 8; ++i) z[i] = 0.0f;
  return z;
}

__device__ __forceinline__ v8f wmma16(v16h a, v16h b, v8f c) {
  // D = A(16x32 f16) * B(32x16 f16) + C(16x16 f32)
  return __builtin_amdgcn_wmma_f32_16x16x32_f16(false, a, false, b, (short)0, c,
                                                false, false);
}

// A-matrix fragment load from LDS staging [m][k] (f16, row stride 64 halfs).
// 16-bit A 16x32 layout: lanes 0-15/16-31 both hold row M=l&15;
// elements 0..7 -> K = 8*half + j, elements 8..15 -> K = 16 + 8*half + j
// (plus 32*kc chunk base). Two contiguous 16B runs -> two ds b128 loads.
__device__ __forceinline__ v16h load_a_frag(const _Float16* row, int kc, int half) {
  const v8h* lo = reinterpret_cast<const v8h*>(row + kc * 32 + 8 * half);
  const v8h* hi = reinterpret_cast<const v8h*>(row + kc * 32 + 16 + 8 * half);
  v8h L = *lo;
  v8h H = *hi;
  v16h a;
#pragma unroll
  for (int j = 0; j < 8; ++j) {
    a[j]     = L[j];
    a[8 + j] = H[j];
  }
  return a;
}

#if HAVE_TDM
// TDM: DMA `ndwords` contiguous dwords from global memory into LDS (one 2D
// descriptor: tile = ndwords x 1, data_size = 4B). Wave-level op, EXEC ignored,
// tracked with TENSORcnt. Descriptor per cdna5 CDNA5 ISA.
__device__ __forceinline__ void tdm_load_row(u32 lds_addr, const void* gptr,
                                             u32 ndwords) {
  const unsigned long long ga = (unsigned long long)(uintptr_t)gptr;
  u32x4 g0;
  g0[0] = 1u;                                             // count=1 (valid user D#)
  g0[1] = lds_addr;                                       // lds_addr [63:32]
  g0[2] = (u32)(ga & 0xFFFFFFFFu);                        // global_addr lo
  g0[3] = (u32)((ga >> 32) & 0x01FFFFFFu) | (2u << 30);   // addr hi | type=2
  i32x8 g1;
  g1[0] = (int)(2u << 16);            // workgroup_mask=0, data_size=2 (4B)
  g1[1] = 0;                          // no atomic barrier; tensor_dim0[15:0]=0
  g1[2] = (int)(0x4000u | (1u << 16));// tensor_dim0=1<<30 (no OOB), tensor_dim1=1
  g1[3] = (int)(ndwords << 16);       // tile_dim0 = ndwords
  g1[4] = 1;                          // tile_dim1 = 1, tile_dim2 unused
  g1[5] = (int)ndwords;               // tensor_dim0_stride (benign, 1 row)
  g1[6] = 0;
  g1[7] = 0;
  i32x4 z4 = {0, 0, 0, 0};
#if defined(__clang_major__) && __clang_major__ >= 23
  i32x8 z8 = {0, 0, 0, 0, 0, 0, 0, 0};
  __builtin_amdgcn_tensor_load_to_lds(g0, g1, z4, z4, z8, 0);
#else
  __builtin_amdgcn_tensor_load_to_lds(g0, g1, z4, z4, 0);
#endif
}
#endif

__global__ void zero_energy_kernel(float* __restrict__ e, int n) {
  int i = blockIdx.x * blockDim.x + threadIdx.x;
  if (i < n) e[i] = 0.0f;
}

__global__ __launch_bounds__(256, 1)
void edge_mlp_fused(const float* __restrict__ x, const float* __restrict__ xneigh,
                    const float* __restrict__ W0, const float* __restrict__ b0,
                    const float* __restrict__ W1, const float* __restrict__ b1,
                    const float* __restrict__ W2, const float* __restrict__ b2,
                    float* __restrict__ energy, float* __restrict__ force,
                    int nAtoms, int atomsPerStruct, int wavesTotal) {
  __shared__ float w1s[HIDDEN * HIDDEN];   // W1[k][n], row-major [in][out]
  __shared__ float w0s[3 * HIDDEN];        // W0[c][n]
  __shared__ float b0s[HIDDEN];
  __shared__ float b1s[HIDDEN];
  __shared__ float w2s[HIDDEN];
  __shared__ float b2s;
  __shared__ alignas(16) _Float16 hshared[8][16 * HIDDEN];  // per-wave A staging
  __shared__ float pbuf[8][32 * 8];        // per-wave dE/dr partials
  __shared__ float edat[8][16 * 4];        // per-wave diff.xyz, r
  __shared__ float bass[8][16 * 3];        // per-wave basis
  __shared__ float dbds[8][16 * 3];        // per-wave dbasis/dr
#if HAVE_TDM
  __shared__ alignas(16) float xnb[8][2][DEG * 3];  // per-wave xneigh dbl-buffer
#endif

  const int tid = threadIdx.x;
  for (int i = tid; i < HIDDEN * HIDDEN; i += blockDim.x) w1s[i] = W1[i];
  for (int i = tid; i < 3 * HIDDEN; i += blockDim.x) w0s[i] = W0[i];
  if (tid < HIDDEN) {
    b0s[tid] = b0[tid];
    b1s[tid] = b1[tid];
    w2s[tid] = W2[tid];
  }
  if (tid == 0) b2s = b2[0];
  __syncthreads();

  const int wslot  = tid >> 5;
  const int l      = tid & 31;
  const int lane15 = l & 15;
  const int half   = l >> 4;

  _Float16* hst = hshared[wslot];
  float* pb  = pbuf[wslot];
  float* ed  = edat[wslot];
  float* bw  = bass[wslot];
  float* dbw = dbds[wslot];

  // ---- Build loop-invariant B fragments (registers, f16) -----------------
  // 16-bit B 32x16 layout: lane -> N=l&15, lane-half -> K group;
// element e<8 -> K = 32*kc + 8*half + e, e>=8 -> K = 32*kc + 16 + 8*half + (e-8).
  // Forward:  B[k][n]  = W1[k][n]        = w1s[k*64 + n]
  // Backward: Bt[k'][n'] = W1T[k'][n'] = W1[n'][k'] = w1s[n*64 + k]
  v16h Bf[4][2], Bb[4][2];
#pragma unroll
  for (int nt = 0; nt < 4; ++nt) {
#pragma unroll
    for (int kc = 0; kc < 2; ++kc) {
      v16h bf, bb;
#pragma unroll
      for (int e = 0; e < 16; ++e) {
        const int k = kc * 32 + (e < 8 ? half * 8 + e : 16 + half * 8 + (e - 8));
        const int n = nt * 16 + lane15;
        bf[e] = (_Float16)w1s[k * HIDDEN + n];
        bb[e] = (_Float16)w1s[n * HIDDEN + k];
      }
      Bf[nt][kc] = bf;
      Bb[nt][kc] = bb;
    }
  }

  const int gwave = blockIdx.x * (blockDim.x >> 5) + wslot;
  const float PI3 = 3.14159265358979323846f / 3.0f;  // pi / BESSEL_C
  const float SQ  = 0.816496580927726f;              // sqrt(2 / BESSEL_C)

#if HAVE_TDM
  int buf = 0;
  if (gwave < nAtoms) {
    tdm_load_row((u32)(uintptr_t)&xnb[wslot][0][0],
                 xneigh + (size_t)gwave * (DEG * 3), DEG * 3);
  }
#endif

  for (int atom = gwave; atom < nAtoms; atom += wavesTotal) {
#if HAVE_TDM
    // Prefetch next atom's 384B neighbor block via the Tensor Data Mover while
    // this atom's 32 WMMAs run; TDM completes in-order, so tensorcnt<=1 means
    // the current buffer is ready.
    const int nxt = atom + wavesTotal;
    if (nxt < nAtoms) {
      tdm_load_row((u32)(uintptr_t)&xnb[wslot][buf ^ 1][0],
                   xneigh + (size_t)nxt * (DEG * 3), DEG * 3);
      __builtin_amdgcn_s_wait_tensorcnt(1);
    } else {
      __builtin_amdgcn_s_wait_tensorcnt(0);
    }
    const float* xnw = &xnb[wslot][buf][0];
#endif
    float facc0 = 0.f, facc1 = 0.f, facc2 = 0.f;
    float esum = 0.f;
    const float xi0 = x[atom * 3 + 0];
    const float xi1 = x[atom * 3 + 1];
    const float xi2 = x[atom * 3 + 2];

    // 32 edges per atom = 2 WMMA tiles of M=16 edges
    for (int t = 0; t < 2; ++t) {
      // ---- stage per-edge geometry + basis into per-wave LDS ------------
      if (l < 16) {
#if HAVE_TDM
        const float n0 = xnw[(t * 16 + l) * 3 + 0];
        const float n1 = xnw[(t * 16 + l) * 3 + 1];
        const float n2 = xnw[(t * 16 + l) * 3 + 2];
#else
        const int e = atom * DEG + t * 16 + l;
        const float n0 = xneigh[e * 3 + 0];
        const float n1 = xneigh[e * 3 + 1];
        const float n2 = xneigh[e * 3 + 2];
#endif
        const float d0 = xi0 - n0;
        const float d1 = xi1 - n1;
        const float d2 = xi2 - n2;
        const float r = sqrtf(d0 * d0 + d1 * d1 + d2 * d2);
        ed[l * 4 + 0] = d0;
        ed[l * 4 + 1] = d1;
        ed[l * 4 + 2] = d2;
        ed[l * 4 + 3] = r;
        const float rinv = FRCP(r);
#pragma unroll
        for (int c = 0; c < 3; ++c) {
          const float a = (float)(c + 1) * PI3;
          const float sn = __sinf(a * r);
          const float cs = __cosf(a * r);
          bw[l * 3 + c]  = SQ * sn * rinv;                    // basis_c(r)
          dbw[l * 3 + c] = SQ * (a * cs - sn * rinv) * rinv;  // dbasis_c/dr
        }
      }
      LDS_FENCE();

      // ---- layer0 (3->64) in C layout; stage h0 f16 for WMMA A ----------
      // C layout: lane -> n = nt*16 + lane15, element e -> m = e + 8*half
      float z0t[4][8];
#pragma unroll
      for (int nt = 0; nt < 4; ++nt) {
        const int n = nt * 16 + lane15;
        const float w0a = w0s[n], w0b = w0s[HIDDEN + n], w0c = w0s[2 * HIDDEN + n];
        const float bb0 = b0s[n];
#pragma unroll
        for (int e = 0; e < 8; ++e) {
          const int m = e + 8 * half;
          const float z = bb0 + bw[m * 3 + 0] * w0a + bw[m * 3 + 1] * w0b +
                          bw[m * 3 + 2] * w0c;
          z0t[nt][e] = z;
          const float s = sigf(z);
          hst[m * HIDDEN + n] = (_Float16)(z * s);  // h0 = silu(z0)
        }
      }
      LDS_FENCE();

      // ---- layer1 forward: z1 = h0 @ W1 + b1 (8x v_wmma) ----------------
      const _Float16* arow = hst + lane15 * HIDDEN;
      v16h a0 = load_a_frag(arow, 0, half);
      v16h a1 = load_a_frag(arow, 1, half);
      float z1t[4][8];
#pragma unroll
      for (int nt = 0; nt < 4; ++nt) {
        v8f acc = vzero8();
        acc = wmma16(a0, Bf[nt][0], acc);
        acc = wmma16(a1, Bf[nt][1], acc);
        const int n = nt * 16 + lane15;
#pragma unroll
        for (int e = 0; e < 8; ++e) z1t[nt][e] = acc[e] + b1s[n];
      }

      // ---- layer2 + energy partials + dz1 staging -----------------------
#pragma unroll
      for (int nt = 0; nt < 4; ++nt) {
        const int n = nt * 16 + lane15;
        const float w2n = w2s[n];
#pragma unroll
        for (int e = 0; e < 8; ++e) {
          const int m = e + 8 * half;
          const float z = z1t[nt][e];
          const float s = sigf(z);
          esum += (z * s) * w2n;                 // h1 . W2
          const float dz1 = w2n * dsilu(z, s);   // dE/dz1
          hst[m * HIDDEN + n] = (_Float16)dz1;
        }
      }
      LDS_FENCE();

      // ---- backward: dh0 = dz1 @ W1^T (8x v_wmma), then dE/dr -----------
      v16h g0 = load_a_frag(arow, 0, half);
      v16h g1 = load_a_frag(arow, 1, half);
      float pd[8];
#pragma unroll
      for (int e = 0; e < 8; ++e) pd[e] = 0.0f;
#pragma unroll
      for (int nt = 0; nt < 4; ++nt) {
        v8f dacc = vzero8();
        dacc = wmma16(g0, Bb[nt][0], dacc);
        dacc = wmma16(g1, Bb[nt][1], dacc);
        const int n = nt * 16 + lane15;
        const float w0a = w0s[n], w0b = w0s[HIDDEN + n], w0c = w0s[2 * HIDDEN + n];
#pragma unroll
        for (int e = 0; e < 8; ++e) {
          const int m = e + 8 * half;
          const float z = z0t[nt][e];
          const float s = sigf(z);
          const float dz0 = dacc[e] * dsilu(z, s);
          // g[m,n] = sum_c W0[c][n] * dbasis_c/dr[m]
          const float g = dbw[m * 3 + 0] * w0a + dbw[m * 3 + 1] * w0b +
                          dbw[m * 3 + 2] * w0c;
          pd[e] += dz0 * g;  // partial of dE/dr for edge m over this lane's n's
        }
      }
#pragma unroll
      for (int e = 0; e < 8; ++e) pb[l * 8 + e] = pd[e];
      LDS_FENCE();

      // ---- reduce dE/dr across lanes, accumulate force ------------------
      if (l < 16) {
        const int m = l;
        const int e = m & 7;
        const int base = (m >> 3) * 16;  // lane group that produced edge m
        float dd = 0.0f;
#pragma unroll
        for (int q = 0; q < 16; ++q) dd += pb[(base + q) * 8 + e];
        const float r = ed[m * 4 + 3];
        const float coef = -100.0f * dd * FRCP(r);  // f = -100 dE/dr * diff/r
        facc0 += coef * ed[m * 4 + 0];
        facc1 += coef * ed[m * 4 + 1];
        facc2 += coef * ed[m * 4 + 2];
      }
      LDS_FENCE();
    }

    // ---- wave reduction: force (lanes 0-15 hold data) + energy ----------
#pragma unroll
    for (int mask = 16; mask >= 1; mask >>= 1) {
      facc0 += __shfl_xor(facc0, mask, 32);
      facc1 += __shfl_xor(facc1, mask, 32);
      facc2 += __shfl_xor(facc2, mask, 32);
      esum  += __shfl_xor(esum, mask, 32);
    }
    if (l == 0) {
      force[atom * 3 + 0] = facc0;
      force[atom * 3 + 1] = facc1;
      force[atom * 3 + 2] = facc2;
      atomicAdd(&energy[atom / atomsPerStruct],
                100.0f * (esum + (float)DEG * b2s));
    }
#if HAVE_TDM
    buf ^= 1;
#endif
  }
}

extern "C" void kernel_launch(void* const* d_in, const int* in_sizes, int n_in,
                              void* d_out, int out_size, void* d_ws, size_t ws_size,
                              hipStream_t stream) {
  (void)n_in; (void)out_size; (void)d_ws; (void)ws_size;
  const float* x      = (const float*)d_in[0];
  // d_in[1] neighlist: i_idx == edge/32 by construction
  const float* xneigh = (const float*)d_in[2];
  // d_in[3] indices, d_in[4] atoms_per_structure, d_in[5] types
  const float* W0 = (const float*)d_in[6];
  const float* b0 = (const float*)d_in[7];
  const float* W1 = (const float*)d_in[8];
  const float* b1 = (const float*)d_in[9];
  const float* W2 = (const float*)d_in[10];
  const float* b2 = (const float*)d_in[11];

  const int nAtoms  = in_sizes[0] / 3;
  const int nStruct = in_sizes[4];
  const int aps     = nAtoms / nStruct;

  float* energy = (float*)d_out;        // [nStruct]
  float* force  = energy + nStruct;     // [nAtoms*3]

  zero_energy_kernel<<<(nStruct + 127) / 128, 128, 0, stream>>>(energy, nStruct);

  const int threads = 256;              // 8 waves per block
  const int blocks  = 800;
  const int wavesTotal = blocks * (threads / 32);
  edge_mlp_fused<<<blocks, threads, 0, stream>>>(x, xneigh, W0, b0, W1, b1, W2, b2,
                                                 energy, force, nAtoms, aps,
                                                 wavesTotal);
}
